// SequentialHRNN_58634893525217
// MI455X (gfx1250) — matrix operand order repaired and checked by reading, not compiled, not run on hardware
//
#include <hip/hip_runtime.h>
#include <hip/hip_bf16.h>
#include <math.h>

// ---------------- problem dims ----------------
#define T_STEPS 512
#define BATCH   64
#define DIN     256
#define HDIM    512
#define DOUT    128

typedef __attribute__((ext_vector_type(16))) __bf16 v16bf;
typedef __attribute__((ext_vector_type(8)))  __bf16 v8bf;
typedef __attribute__((ext_vector_type(8)))  float  v8f;

// ---------------- packed-weight geometry ----------------
#define L0_NKT 40   // K = 256 (x) + 512 (h1) + 512 (h0) = 1280
#define L1_NKT 48   // K = 512*3
#define L2_NKT 32   // K = 512*2
#define WC_NKT 16   // K = 512
#define L0_SZ (32 * L0_NKT * 512)
#define L1_SZ (32 * L1_NKT * 512)
#define L2_SZ (32 * L2_NKT * 512)
#define WC_SZ (8  * WC_NKT * 512)
#define PACK_TOTAL (L0_SZ + L1_SZ + L2_SZ + WC_SZ)   // 2,031,616 bf16

// ---------------- workspace layout (bytes) ----------------
#define BAR_OFF    0
#define HBUF_OFF   256
#define HBUF_BYTES (2 * 3 * BATCH * HDIM * 2)
#define XBF_OFF    (HBUF_OFF + HBUF_BYTES)
#define XBF_BYTES  ((size_t)T_STEPS * BATCH * DIN * 2)
#define WPK_OFF    (XBF_OFF + XBF_BYTES)
#define PRED_ELEMS ((size_t)T_STEPS * BATCH * DOUT)

// LDS A-chunk staging: 64 rows x 128 cols, pitch padded to 136 bf16 (rows shift 4 banks)
#define CH_COLS  128
#define CH_PITCH 136

// ================= preamble: pack weights to bf16 fragment order =================
__global__ __launch_bounds__(256) void hrnn_pack(
    const float* __restrict__ Wbu0, const float* __restrict__ Wtd0, const float* __restrict__ Wrec0,
    const float* __restrict__ Wbu1, const float* __restrict__ Wtd1, const float* __restrict__ Wrec1,
    const float* __restrict__ Wbu2, const float* __restrict__ Wrec2, const float* __restrict__ Wc,
    unsigned short* __restrict__ wp_u)
{
  long idx = (long)blockIdx.x * blockDim.x + threadIdx.x;
  if (idx >= (long)PACK_TOTAL) return;
  int L; long rel; int nkt;
  if (idx < L0_SZ)                      { L = 0; rel = idx;                         nkt = L0_NKT; }
  else if (idx < L0_SZ + L1_SZ)         { L = 1; rel = idx - L0_SZ;                 nkt = L1_NKT; }
  else if (idx < L0_SZ + L1_SZ + L2_SZ) { L = 2; rel = idx - L0_SZ - L1_SZ;         nkt = L2_NKT; }
  else                                  { L = 3; rel = idx - L0_SZ - L1_SZ - L2_SZ; nkt = WC_NKT; }
  int per   = nkt * 512;
  int strip = (int)(rel / per);
  int rem   = (int)(rel % per);
  int kt    = rem >> 9;
  int r3    = rem & 511;
  int lane  = r3 >> 4;
  int j     = r3 & 15;
  int k = kt * 32 + ((lane >> 4) << 3) + (j & 7) + ((j >> 3) << 4);   // B-fragment: lane half h covers K [8h,8h+8) and [16+8h,24+8h)
  int n = strip * 16 + (lane & 15);
  float v;
  if (L == 0)      v = (k < 256)  ? Wbu0[k * 512 + n]
                     : (k < 768)  ? Wtd0[(k - 256) * 512 + n]
                                  : Wrec0[(k - 768) * 512 + n];
  else if (L == 1) v = (k < 512)  ? Wbu1[k * 512 + n]
                     : (k < 1024) ? Wtd1[(k - 512) * 512 + n]
                                  : Wrec1[(k - 1024) * 512 + n];
  else if (L == 2) v = (k < 512)  ? Wbu2[k * 512 + n]
                                  : Wrec2[(k - 512) * 512 + n];
  else             v = Wc[k * 128 + n];
  reinterpret_cast<__bf16*>(wp_u)[idx] = (__bf16)v;
}

// ================= preamble: x fp32 -> bf16 =================
__global__ __launch_bounds__(256) void hrnn_cvt_x(const float* __restrict__ x,
                                                  unsigned short* __restrict__ xb, long n)
{
  long i = (long)blockIdx.x * blockDim.x + threadIdx.x;
  if (i < n) reinterpret_cast<__bf16*>(xb)[i] = (__bf16)x[i];
}

// ================= grid-wide sense barrier (device-scope atomics) =================
__device__ __forceinline__ void grid_barrier(unsigned* bar)
{
  __threadfence();
  __syncthreads();
  if (threadIdx.x == 0) {
    unsigned gen = __hip_atomic_load(bar + 1, __ATOMIC_RELAXED, __HIP_MEMORY_SCOPE_AGENT);
    unsigned arr = __hip_atomic_fetch_add(bar, 1u, __ATOMIC_ACQ_REL, __HIP_MEMORY_SCOPE_AGENT);
    if (arr == (unsigned)(gridDim.x - 1)) {
      __hip_atomic_store(bar, 0u, __ATOMIC_RELAXED, __HIP_MEMORY_SCOPE_AGENT);
      __hip_atomic_fetch_add(bar + 1, 1u, __ATOMIC_RELEASE, __HIP_MEMORY_SCOPE_AGENT);
    } else {
      while (__hip_atomic_load(bar + 1, __ATOMIC_ACQUIRE, __HIP_MEMORY_SCOPE_AGENT) == gen)
        __builtin_amdgcn_s_sleep(2);
    }
  }
  __syncthreads();
  __threadfence();
}

// ================= async stage of one 64x128 A-chunk into LDS =================
// Per wave: 16 rows, 2 rows per instruction (32 lanes x 16B = 512B).
__device__ __forceinline__ void stage_chunk_async(const __bf16* __restrict__ src, int pitch,
                                                  int colbase, __bf16* dst, int wave, int lane)
{
  const int sub = lane >> 4;            // row parity within pair
  const int bo  = (lane & 15) * 8;      // element offset (16B granules)
#pragma unroll
  for (int i = 0; i < 8; ++i) {
    int row = wave * 16 + i * 2 + sub;
    const __bf16* g = src + (size_t)row * pitch + colbase + bo;
    unsigned           lofs = (unsigned)(size_t)(dst + row * CH_PITCH + bo);  // flat->LDS: addr[31:0]
    unsigned long long ga   = (unsigned long long)(size_t)g;
    asm volatile("global_load_async_to_lds_b128 %0, %1, off"
                 :: "v"(lofs), "v"(ga) : "memory");
  }
}

// ================= WMMA over one staged LDS chunk (4 K-tiles) =================
// A layout (ISA 16-bit A 16x32): lane half -> kb=(l>>4)*8; K in [kb,kb+8) and [kb+16,kb+24)
__device__ __forceinline__ void accum_chunk_lds(const __bf16* lb, const __bf16*& wk,
                                                int lane, v8f* acc)
{
  const int nlo   = lane & 15;
  const int khalf = (lane >> 4) << 3;
#pragma unroll
  for (int kt = 0; kt < 4; ++kt) {
    v16bf b = *(const v16bf*)(wk + (size_t)lane * 16);
    __builtin_prefetch(wk + 1024 + (size_t)lane * 16, 0, 1);
    wk += 512;
    const __bf16* ap = lb + nlo * CH_PITCH + kt * 32 + khalf;
#pragma unroll
    for (int r = 0; r < 4; ++r) {
      const __bf16* arp = ap + r * 16 * CH_PITCH;
      v8bf lo = *(const v8bf*)(arp);
      v8bf hi = *(const v8bf*)(arp + 16);
      v16bf a = __builtin_shufflevector(lo, hi, 0,1,2,3,4,5,6,7,8,9,10,11,12,13,14,15);
      acc[r] = __builtin_amdgcn_wmma_f32_16x16x32_bf16(false, a, false, b,
                                                       (short)0, acc[r], false, false);
    }
  }
}

// chunk index -> (source, pitch, column base); wave-uniform branches
__device__ __forceinline__ const __bf16* chunk_src(int layer, int c, const __bf16* xt,
                                                   const __bf16* h0, const __bf16* h1,
                                                   const __bf16* h2, int& pitch, int& colbase)
{
  if (layer == 0) {
    if (c < 2)      { pitch = DIN;  colbase = c * CH_COLS;       return xt; }
    else if (c < 6) { pitch = HDIM; colbase = (c - 2) * CH_COLS; return h1; }
    else            { pitch = HDIM; colbase = (c - 6) * CH_COLS; return h0; }
  } else if (layer == 1) {
    pitch = HDIM; colbase = (c & 3) * CH_COLS;
    return (c < 4) ? h0 : (c < 8) ? h2 : h1;
  } else {
    pitch = HDIM; colbase = (c & 3) * CH_COLS;
    return (c < 4) ? h1 : h2;
  }
}

// ================= persistent recurrence kernel =================
// 24 WGs x 4 waves = 96 waves = one wave per (layer, 16-col strip); layer uniform per WG.
__global__ __launch_bounds__(128) void hrnn_persistent(
    const float* __restrict__ b0, const float* __restrict__ b1, const float* __restrict__ b2,
    const unsigned short* __restrict__ xbf_u,
    const unsigned short* __restrict__ wpk_u,
    unsigned short* __restrict__ hbuf_u,
    float* __restrict__ topOut,
    unsigned* __restrict__ bar)
{
  __shared__ __align__(16) __bf16 sbuf[2][64 * CH_PITCH];   // 2 x 17KB A-chunk buffers

  const __bf16* xbf  = reinterpret_cast<const __bf16*>(xbf_u);
  const __bf16* wpk  = reinterpret_cast<const __bf16*>(wpk_u);
  __bf16*       hbuf = reinterpret_cast<__bf16*>(hbuf_u);

  const int lane  = threadIdx.x & 31;
  const int wave  = threadIdx.x >> 5;
  const int strip = blockIdx.x * 4 + wave;     // 0..95
  const int layer = strip >> 5;
  const int sl    = strip & 31;
  const int n0    = sl * 16;
  const int nlo   = lane & 15;

  const int    nkt  = (layer == 0) ? L0_NKT : (layer == 1) ? L1_NKT : L2_NKT;
  const int    nch  = nkt / 4;                 // 10 / 12 / 8 chunks of 4 K-tiles
  const size_t loff = (layer == 0) ? 0
                    : (layer == 1) ? (size_t)L0_SZ
                                   : (size_t)L0_SZ + (size_t)L1_SZ;
  const __bf16* wp = wpk + loff + (size_t)sl * nkt * 512;

  const float* bptr = (layer == 0) ? b0 : (layer == 1) ? b1 : b2;
  const float  bias = bptr[n0 + nlo];

  for (int t = 0; t < T_STEPS; ++t) {
    const int cur = t & 1, nxt = cur ^ 1;
    const __bf16* h0 = hbuf + (size_t)(cur * 3 + 0) * BATCH * HDIM;
    const __bf16* h1 = hbuf + (size_t)(cur * 3 + 1) * BATCH * HDIM;
    const __bf16* h2 = hbuf + (size_t)(cur * 3 + 2) * BATCH * HDIM;
    const __bf16* xt = xbf  + (size_t)t * BATCH * DIN;

    // prologue: stage chunk 0
    {
      int p, cb;
      const __bf16* s = chunk_src(layer, 0, xt, h0, h1, h2, p, cb);
      stage_chunk_async(s, p, cb, &sbuf[0][0], wave, lane);
    }
    asm volatile("s_wait_asynccnt 0x0" ::: "memory");
    __syncthreads();

    v8f acc[4] = {};
    const __bf16* wk = wp;
    for (int c = 0; c < nch; ++c) {
      if (c + 1 < nch) {                      // overlap: stage next chunk while computing
        int p, cb;
        const __bf16* s = chunk_src(layer, c + 1, xt, h0, h1, h2, p, cb);
        stage_chunk_async(s, p, cb, &sbuf[(c + 1) & 1][0], wave, lane);
      }
      accum_chunk_lds(&sbuf[c & 1][0], wk, lane, acc);
      asm volatile("s_wait_asynccnt 0x0" ::: "memory");
      __syncthreads();
    }

    // epilogue: bias + tanh, write next-step hidden (bf16) and top output (fp32)
    __bf16* hn = hbuf + (size_t)(nxt * 3 + layer) * BATCH * HDIM;
#pragma unroll
    for (int r = 0; r < 4; ++r) {
#pragma unroll
      for (int e = 0; e < 8; ++e) {
        float v   = tanhf(acc[r][e] + bias);
        int   row = r * 16 + e + ((lane >> 4) << 3);
        int   col = n0 + nlo;
        hn[(size_t)row * HDIM + col] = (__bf16)v;
        if (layer == 2)
          topOut[((size_t)t * BATCH + row) * HDIM + col] = v;
      }
    }
    grid_barrier(bar);
  }
}

// ================= final projection: pred = top @ Wc + bc =================
__global__ __launch_bounds__(256) void hrnn_pred(
    const float* __restrict__ top,
    const unsigned short* __restrict__ wcp_u,
    const float* __restrict__ bc,
    float* __restrict__ pred)
{
  const __bf16* wcp = reinterpret_cast<const __bf16*>(wcp_u);
  const int lane = threadIdx.x & 31;
  const int gw   = blockIdx.x * (blockDim.x >> 5) + (threadIdx.x >> 5);
  const int nlo  = lane & 15;
  const int kh   = (lane >> 4) << 3;
  const int m0   = gw * 16;

  v8f acc[8] = {};
  for (int kt = 0; kt < WC_NKT; ++kt) {
    const float* ap = top + (size_t)(m0 + nlo) * HDIM + kt * 32 + kh;
    v8f f0 = *(const v8f*)(ap);
    v8f f1 = *(const v8f*)(ap + 16);
    v16bf a = __builtin_shufflevector(__builtin_convertvector(f0, v8bf),
                                      __builtin_convertvector(f1, v8bf),
                                      0,1,2,3,4,5,6,7,8,9,10,11,12,13,14,15);
#pragma unroll
    for (int nt = 0; nt < 8; ++nt) {
      v16bf b = *(const v16bf*)(wcp + ((size_t)(nt * WC_NKT + kt) * 32 + lane) * 16);
      acc[nt] = __builtin_amdgcn_wmma_f32_16x16x32_bf16(false, a, false, b,
                                                        (short)0, acc[nt], false, false);
    }
  }
#pragma unroll
  for (int nt = 0; nt < 8; ++nt) {
#pragma unroll
    for (int e = 0; e < 8; ++e) {
      int n   = nt * 16 + nlo;
      int row = m0 + e + ((lane >> 4) << 3);
      pred[(size_t)row * DOUT + n] = acc[nt][e] + bc[n];
    }
  }
}

// ================= host launcher =================
extern "C" void kernel_launch(void* const* d_in, const int* in_sizes, int n_in,
                              void* d_out, int out_size, void* d_ws, size_t ws_size,
                              hipStream_t stream)
{
  (void)in_sizes; (void)n_in; (void)out_size; (void)ws_size;

  const float* x     = (const float*)d_in[0];
  const float* Wbu0  = (const float*)d_in[1];
  const float* Wtd0  = (const float*)d_in[2];
  const float* Wrec0 = (const float*)d_in[3];
  const float* b0    = (const float*)d_in[4];
  const float* Wbu1  = (const float*)d_in[5];
  const float* Wtd1  = (const float*)d_in[6];
  const float* Wrec1 = (const float*)d_in[7];
  const float* b1    = (const float*)d_in[8];
  const float* Wbu2  = (const float*)d_in[9];
  const float* Wrec2 = (const float*)d_in[10];
  const float* b2    = (const float*)d_in[11];
  const float* Wc    = (const float*)d_in[12];
  const float* bc    = (const float*)d_in[13];

  char* ws = (char*)d_ws;
  unsigned*       bar  = (unsigned*)(ws + BAR_OFF);
  unsigned short* hbuf = (unsigned short*)(ws + HBUF_OFF);
  unsigned short* xbf  = (unsigned short*)(ws + XBF_OFF);
  unsigned short* wpk  = (unsigned short*)(ws + WPK_OFF);
  unsigned short* wcp  = wpk + (L0_SZ + L1_SZ + L2_SZ);

  float* pred = (float*)d_out;
  float* topO = (float*)d_out + PRED_ELEMS;

  hipMemsetAsync(d_ws, 0, HBUF_OFF + HBUF_BYTES, stream);

  hrnn_pack<<<(PACK_TOTAL + 255) / 256, 256, 0, stream>>>(
      Wbu0, Wtd0, Wrec0, Wbu1, Wtd1, Wrec1, Wbu2, Wrec2, Wc, wpk);
  long xn = (long)T_STEPS * BATCH * DIN;
  hrnn_cvt_x<<<(int)((xn + 255) / 256), 256, 0, stream>>>(x, xbf, xn);

  hrnn_persistent<<<24, 128, 0, stream>>>(b0, b1, b2, xbf, wpk, hbuf, topO, bar);

  hrnn_pred<<<256, 256, 0, stream>>>(topO, wcp, bc, pred);
}
